// DirectionalAGLGF_79027398246895
// MI455X (gfx1250) — matrix operand order repaired and checked by reading, not compiled, not run on hardware
//
#include <hip/hip_runtime.h>
#include <hip/hip_bf16.h>

#define LSEQ   4096
#define BATCHN 2
#define DIN    256
#define DMODEL 128
#define NSTATE 16
#define NPROJ  64   // dt_rank(8) + 2*N(32) = 40, padded to 64 -> N multiple of 32
#define DTK    32   // dt_rank 8 padded to one K=32 WMMA step

typedef __attribute__((ext_vector_type(16))) _Float16 v16h;
typedef __attribute__((ext_vector_type(8)))  _Float16 v8h;
typedef __attribute__((ext_vector_type(8)))  float    v8f;

// ---------------------------------------------------------------------------
// B-operand swizzle: weights are pre-packed so that a wave's B fragment for a
// 16x16x32 WMMA tile is a contiguous 32B run per lane:
//   tile  = (n/16) * (K/32) + (k/32)           (K-minor: streams through K)
//   entry = tile*512 + (lane = (n&15) + 16*((k>>4)&1))*16 + (k&15)
// so a lane reads b[e] = B[k0 + kh*16 + e][n] from one v16h load.
// ---------------------------------------------------------------------------
__device__ __forceinline__ int bswz(int k, int n, int K) {
  return (((n >> 4) * (K >> 5) + (k >> 5)) << 9) +
         (((n & 15) + (((k >> 4) & 1) << 4)) << 4) + (k & 15);
}

// ---------------------------------------------------------------------------
// WMMA GEMM: C[M,N] = A[M,KC] (f16 row-major) x Bsw (f16 pre-swizzled).
// One wave computes a 16(M) x 32(N) tile: 2 WMMAs per K-step sharing the A
// fragment. KC is compile-time so the K loop fully unrolls.
// A fragment layout (wave32): lane = M + 16*kh; lane holds two contiguous
// 8-f16 runs at k = kh*16 and k = kh*16 + 8 -> two b128 loads.
// C/D f32 layout: lane = N + 16*half, VGPR r -> M = r + 8*half.
// ---------------------------------------------------------------------------
template <int KC, bool TRANSPOSED, bool HASBIAS>
__global__ void wmma_gemm_kernel(const _Float16* __restrict__ A,
                                 const _Float16* __restrict__ Bsw,
                                 float* __restrict__ C,
                                 const float* __restrict__ bias,
                                 int M, int N) {
  constexpr int KTILES = KC / 32;
  const int lane  = threadIdx.x & 31;
  const int wave  = (blockIdx.x * blockDim.x + threadIdx.x) >> 5;
  const int npair = N >> 5;                 // # of 32-wide N tile pairs
  const int tm    = (wave / npair) << 4;
  const int nt0   = (wave % npair) << 1;    // first 16-wide N-tile index
  const int kh    = lane >> 4;

  const _Float16* arow = A + (size_t)(tm + (lane & 15)) * KC + kh * 16;
  const _Float16* b0   = Bsw + ((size_t)nt0 * KTILES << 9) + lane * 16;
  const _Float16* b1   = b0 + ((size_t)KTILES << 9);

  v8f c0 = {0.f, 0.f, 0.f, 0.f, 0.f, 0.f, 0.f, 0.f};
  v8f c1 = {0.f, 0.f, 0.f, 0.f, 0.f, 0.f, 0.f, 0.f};
#pragma unroll
  for (int kt = 0; kt < KTILES; ++kt) {
    if (kt + 1 < KTILES) {  // stream next B panels toward L0 (global_prefetch_b8)
      __builtin_prefetch((const void*)(b0 + ((kt + 1) << 9)), 0, 1);
      __builtin_prefetch((const void*)(b1 + ((kt + 1) << 9)), 0, 1);
    }
    const v8h alo = *(const v8h*)(arow + kt * 32);
    const v8h ahi = *(const v8h*)(arow + kt * 32 + 8);
    const v16h a  = __builtin_shufflevector(alo, ahi, 0, 1, 2, 3, 4, 5, 6, 7,
                                            8, 9, 10, 11, 12, 13, 14, 15);
    const v16h bb0 = *(const v16h*)(b0 + (kt << 9));
    const v16h bb1 = *(const v16h*)(b1 + (kt << 9));
    c0 = __builtin_amdgcn_wmma_f32_16x16x32_f16(false, a, false, bb0, (short)0,
                                                c0, false, false);
    c1 = __builtin_amdgcn_wmma_f32_16x16x32_f16(false, a, false, bb1, (short)0,
                                                c1, false, false);
  }

  const int n0    = (nt0 << 4) + (lane & 15);
  const int mbase = tm + (kh << 3);
#pragma unroll
  for (int r = 0; r < 8; ++r) {
    const int mm = mbase + r;
    float v0 = c0[r];
    float v1 = c1[r];
    if constexpr (HASBIAS) { v0 += bias[n0]; v1 += bias[n0 + 16]; }
    if constexpr (!TRANSPOSED) {
      C[(size_t)mm * N + n0]      = v0;
      C[(size_t)mm * N + n0 + 16] = v1;
    } else {  // mm = b*LSEQ + l  ->  C[b, n, l]  (NCHW output)
      const size_t bi = (size_t)(mm >> 12);
      const int l = mm & (LSEQ - 1);
      C[(bi * N + n0) * LSEQ + l]      = v0;
      C[(bi * N + n0 + 16) * LSEQ + l] = v1;
    }
  }
}

// ---------------------------------------------------------------------------
// Weight prep: cast to f16 into the WMMA-swizzled B layout, zero-padding
// N=40->64 (x_proj) and K=8->32 (dt).
// ---------------------------------------------------------------------------
__global__ void prep_weights_kernel(const float* __restrict__ w_in_x,
                                    const float* __restrict__ w_in_z,
                                    const float* __restrict__ xp_f,
                                    const float* __restrict__ xp_b,
                                    const float* __restrict__ dtw_f,
                                    const float* __restrict__ dtw_b,
                                    const float* __restrict__ out_w,
                                    _Float16* __restrict__ Bx, _Float16* __restrict__ Bz,
                                    _Float16* __restrict__ Bpf, _Float16* __restrict__ Bpb,
                                    _Float16* __restrict__ Bdf, _Float16* __restrict__ Bdb,
                                    _Float16* __restrict__ Bo) {
  __builtin_amdgcn_s_wait_tensorcnt(0);  // CDNA5 split-counter path (no-op wait)
  const int i = blockIdx.x * blockDim.x + threadIdx.x;
  if (i < DMODEL * DIN) {                       // Bx/Bz: [K=128, N=256] from (256,128)
    const int k = i / DIN, n = i % DIN;
    const int s = bswz(k, n, DMODEL);
    Bx[s] = (_Float16)w_in_x[n * DMODEL + k];
    Bz[s] = (_Float16)w_in_z[n * DMODEL + k];
  }
  if (i < DIN * NPROJ) {                        // Bp*: [K=256, N=64] from (40,256), pad N
    const int k = i / NPROJ, n = i % NPROJ;
    const int s = bswz(k, n, DIN);
    Bpf[s] = (_Float16)((n < 40) ? xp_f[n * DIN + k] : 0.f);
    Bpb[s] = (_Float16)((n < 40) ? xp_b[n * DIN + k] : 0.f);
  }
  if (i < DTK * DIN) {                          // Bd*: [K=32, N=256] from (256,8), pad K
    const int k = i / DIN, n = i % DIN;
    const int s = bswz(k, n, DTK);
    Bdf[s] = (_Float16)((k < 8) ? dtw_f[n * 8 + k] : 0.f);
    Bdb[s] = (_Float16)((k < 8) ? dtw_b[n * 8 + k] : 0.f);
  }
  if (i < DIN * DMODEL) {                       // Bo: [K=256, N=128] from (128,256)
    const int k = i / DMODEL, n = i % DMODEL;
    Bo[bswz(k, n, DIN)] = (_Float16)out_w[n * DIN + k];
  }
}

// ---------------------------------------------------------------------------
// Dual layernorm over C=128, reading NCHW (B,C,L) and writing (B*L, C) f16.
// ---------------------------------------------------------------------------
__device__ __forceinline__ float block_sum_128(float v, float* sh) {
#pragma unroll
  for (int o = 16; o > 0; o >>= 1) v += __shfl_xor(v, o, 32);
  const int w = threadIdx.x >> 5;
  if ((threadIdx.x & 31) == 0) sh[w] = v;
  __syncthreads();
  const float r = sh[0] + sh[1] + sh[2] + sh[3];
  __syncthreads();
  return r;
}

__global__ void layernorm2_kernel(const float* __restrict__ x1, const float* __restrict__ x2,
                                  const float* __restrict__ wq,  const float* __restrict__ bq,
                                  const float* __restrict__ wkv, const float* __restrict__ bkv,
                                  _Float16* __restrict__ x1n, _Float16* __restrict__ x2n) {
  __shared__ float sh[4];
  const int row = blockIdx.x;           // b*LSEQ + l
  const int b   = row >> 12;
  const int l   = row & (LSEQ - 1);
  const int c   = threadIdx.x;          // 0..127
  const size_t src = ((size_t)(b * DMODEL + c)) * LSEQ + l;
  const float v1 = x1[src];
  const float v2 = x2[src];
  const float s1 = block_sum_128(v1, sh);
  const float q1 = block_sum_128(v1 * v1, sh);
  const float s2 = block_sum_128(v2, sh);
  const float q2 = block_sum_128(v2 * v2, sh);
  const float inv = 1.f / (float)DMODEL;
  const float mu1 = s1 * inv, mu2 = s2 * inv;
  const float r1 = rsqrtf(q1 * inv - mu1 * mu1 + 1e-5f);
  const float r2 = rsqrtf(q2 * inv - mu2 * mu2 + 1e-5f);
  x1n[(size_t)row * DMODEL + c] = (_Float16)((v1 - mu1) * r1 * wq[c]  + bq[c]);
  x2n[(size_t)row * DMODEL + c] = (_Float16)((v2 - mu2) * r2 * wkv[c] + bkv[c]);
}

// ---------------------------------------------------------------------------
// Depthwise causal conv (K=4) + SiLU. FLIP=true processes the sequence
// reversed (buffers stay in flipped logical order; un-flipped at the gate).
// ---------------------------------------------------------------------------
template <bool FLIP>
__global__ void conv_silu_kernel(const float* __restrict__ xin, const float* __restrict__ w,
                                 const float* __restrict__ bias,
                                 float* __restrict__ xc, _Float16* __restrict__ xch) {
  const int idx = blockIdx.x * blockDim.x + threadIdx.x;  // (b*L + t)*256 + d
  const int d = idx & (DIN - 1);
  const int t = (idx >> 8) & (LSEQ - 1);
  const int b = idx >> 20;
  float acc = bias[d];
#pragma unroll
  for (int k = 0; k < 4; ++k) {
    const int j = t - 3 + k;
    if (j >= 0) {
      const int p = FLIP ? (LSEQ - 1 - j) : j;
      acc += w[d * 4 + k] * xin[(((size_t)b * LSEQ + p) << 8) + d];
    }
  }
  const float s = acc / (1.f + __expf(-acc));   // SiLU
  xc[idx]  = s;
  xch[idx] = (_Float16)s;
}

// dt_r (first 8 cols of dbl) -> f16 [rows,32], zero-padded K for the dt GEMM
__global__ void dtprep_kernel(const float* __restrict__ dbl, _Float16* __restrict__ dtr) {
  const int idx = blockIdx.x * blockDim.x + threadIdx.x;
  const int c   = idx & 31;
  const int row = idx >> 5;
  dtr[idx] = (c < 8) ? (_Float16)dbl[(size_t)row * NPROJ + c] : (_Float16)0.f;
}

// ---------------------------------------------------------------------------
// Selective scan: one channel (b,d) per 16-lane half-wave; lane n holds state
// h_n. y_t = sum_n h_n*C_n reduced with 4x shfl_xor inside the half-wave.
// grid: 32 blocks x 256 thr = 512 channels x 16 states.
// ---------------------------------------------------------------------------
__global__ void ssm_scan_kernel(const float* __restrict__ u, const float* __restrict__ dtraw,
                                const float* __restrict__ dbl, const float* __restrict__ dt_bias,
                                const float* __restrict__ A_log, const float* __restrict__ Dp,
                                float* __restrict__ y) {
  const int n  = threadIdx.x & 15;
  const int ch = blockIdx.x * (blockDim.x >> 4) + (threadIdx.x >> 4);
  const int b  = ch >> 8;
  const int d  = ch & (DIN - 1);
  const float A   = -__expf(A_log[d * NSTATE + n]);
  const float dtb = dt_bias[d];
  const float Dd  = Dp[d];
  float h = 0.f;
  const size_t rbase = (size_t)b * LSEQ;
  for (int t = 0; t < LSEQ; ++t) {
    const size_t row = rbase + t;
    float dtv = dtraw[(row << 8) + d] + dtb;
    dtv = (dtv > 20.f) ? dtv : log1pf(__expf(dtv));        // softplus
    const float uv = u[(row << 8) + d];
    const float Bn = dbl[(row << 6) + 8 + n];
    const float Cn = dbl[(row << 6) + 24 + n];
    h = __expf(dtv * A) * h + (dtv * uv) * Bn;
    float p = h * Cn;
    p += __shfl_xor(p, 1, 32);
    p += __shfl_xor(p, 2, 32);
    p += __shfl_xor(p, 4, 32);
    p += __shfl_xor(p, 8, 32);
    if (n == 0) y[(row << 8) + d] = p + uv * Dd;
  }
}

// (y_f + flip(y_b)) * silu(z) -> f16 for the out GEMM
__global__ void gate_kernel(const float* __restrict__ yf, const float* __restrict__ yb,
                            const float* __restrict__ z, _Float16* __restrict__ yg) {
  const int idx = blockIdx.x * blockDim.x + threadIdx.x;
  const int d = idx & (DIN - 1);
  const int t = (idx >> 8) & (LSEQ - 1);
  const int b = idx >> 20;
  const float ybv = yb[(((size_t)b * LSEQ + (LSEQ - 1 - t)) << 8) + d];
  const float zv  = z[idx];
  const float g   = zv / (1.f + __expf(-zv));
  yg[idx] = (_Float16)((yf[idx] + ybv) * g);
}

// ---------------------------------------------------------------------------
extern "C" void kernel_launch(void* const* d_in, const int* in_sizes, int n_in,
                              void* d_out, int out_size, void* d_ws, size_t ws_size,
                              hipStream_t stream) {
  const float* x1       = (const float*)d_in[0];
  const float* x2       = (const float*)d_in[1];
  const float* ln_q_w   = (const float*)d_in[2];
  const float* ln_q_b   = (const float*)d_in[3];
  const float* ln_kv_w  = (const float*)d_in[4];
  const float* ln_kv_b  = (const float*)d_in[5];
  const float* w_in_x   = (const float*)d_in[6];
  const float* w_in_z   = (const float*)d_in[7];
  const float* conv_w   = (const float*)d_in[8];
  const float* conv_b   = (const float*)d_in[9];
  const float* conv_w_b = (const float*)d_in[10];
  const float* conv_b_b = (const float*)d_in[11];
  const float* x_proj_w   = (const float*)d_in[12];
  const float* x_proj_w_b = (const float*)d_in[13];
  const float* dt_w   = (const float*)d_in[14];
  const float* dt_b   = (const float*)d_in[15];
  const float* dt_w_b = (const float*)d_in[16];
  const float* dt_b_b = (const float*)d_in[17];
  const float* A_log   = (const float*)d_in[18];
  const float* A_log_b = (const float*)d_in[19];
  const float* Dp   = (const float*)d_in[20];
  const float* Dp_b = (const float*)d_in[21];
  const float* out_w = (const float*)d_in[22];
  const float* out_b = (const float*)d_in[23];
  (void)in_sizes; (void)n_in; (void)out_size; (void)ws_size;

  char* base = (char*)d_ws;
  size_t off = 0;
  auto alloc = [&](size_t bytes) -> char* {
    char* p = base + off;
    off = (off + bytes + 255) & ~(size_t)255;
    return p;
  };
  const size_t ML = (size_t)BATCHN * LSEQ;  // 8192 rows

  _Float16* Bx   = (_Float16*)alloc((size_t)DMODEL * DIN * 2);
  _Float16* Bz   = (_Float16*)alloc((size_t)DMODEL * DIN * 2);
  _Float16* Bpf  = (_Float16*)alloc((size_t)DIN * NPROJ * 2);
  _Float16* Bpb  = (_Float16*)alloc((size_t)DIN * NPROJ * 2);
  _Float16* Bdf  = (_Float16*)alloc((size_t)DTK * DIN * 2);
  _Float16* Bdb  = (_Float16*)alloc((size_t)DTK * DIN * 2);
  _Float16* Bo   = (_Float16*)alloc((size_t)DIN * DMODEL * 2);
  _Float16* x1n  = (_Float16*)alloc(ML * DMODEL * 2);
  _Float16* x2n  = (_Float16*)alloc(ML * DMODEL * 2);
  float*    xbuf = (float*)   alloc(ML * DIN * 4);
  float*    zbuf = (float*)   alloc(ML * DIN * 4);
  float*    xc_f = (float*)   alloc(ML * DIN * 4);
  float*    xc_b = (float*)   alloc(ML * DIN * 4);
  _Float16* xcfh = (_Float16*)alloc(ML * DIN * 2);
  _Float16* xcbh = (_Float16*)alloc(ML * DIN * 2);
  float*    dblf = (float*)   alloc(ML * NPROJ * 4);
  float*    dblb = (float*)   alloc(ML * NPROJ * 4);
  _Float16* dtrf = (_Float16*)alloc(ML * DTK * 2);
  _Float16* dtrb = (_Float16*)alloc(ML * DTK * 2);
  float*    dtwf = (float*)   alloc(ML * DIN * 4);
  float*    dtwb = (float*)   alloc(ML * DIN * 4);
  float*    y_f  = (float*)   alloc(ML * DIN * 4);
  float*    y_b  = (float*)   alloc(ML * DIN * 4);
  _Float16* ygh  = (_Float16*)alloc(ML * DIN * 2);

  // waves = (M/16)*(N/32); 8 waves per 256-thread block (all counts divide)
  #define GEMM_GRID(M, N) (((M) >> 4) * ((N) >> 5) / 8)

  prep_weights_kernel<<<128, 256, 0, stream>>>(w_in_x, w_in_z, x_proj_w, x_proj_w_b,
                                               dt_w, dt_w_b, out_w,
                                               Bx, Bz, Bpf, Bpb, Bdf, Bdb, Bo);
  layernorm2_kernel<<<(int)ML, DMODEL, 0, stream>>>(x1, x2, ln_q_w, ln_q_b,
                                                    ln_kv_w, ln_kv_b, x1n, x2n);

  wmma_gemm_kernel<DMODEL, false, false><<<GEMM_GRID((int)ML, DIN), 256, 0, stream>>>(
      x1n, Bx, xbuf, nullptr, (int)ML, DIN);                    // x = x1n @ Wx^T
  wmma_gemm_kernel<DMODEL, false, false><<<GEMM_GRID((int)ML, DIN), 256, 0, stream>>>(
      x2n, Bz, zbuf, nullptr, (int)ML, DIN);                    // z = x2n @ Wz^T

  const int ew_blocks = (int)(ML * DIN / 256);
  conv_silu_kernel<false><<<ew_blocks, 256, 0, stream>>>(xbuf, conv_w,   conv_b,   xc_f, xcfh);
  conv_silu_kernel<true ><<<ew_blocks, 256, 0, stream>>>(xbuf, conv_w_b, conv_b_b, xc_b, xcbh);

  wmma_gemm_kernel<DIN, false, false><<<GEMM_GRID((int)ML, NPROJ), 256, 0, stream>>>(
      xcfh, Bpf, dblf, nullptr, (int)ML, NPROJ);                // dbl fwd
  wmma_gemm_kernel<DIN, false, false><<<GEMM_GRID((int)ML, NPROJ), 256, 0, stream>>>(
      xcbh, Bpb, dblb, nullptr, (int)ML, NPROJ);                // dbl bwd

  dtprep_kernel<<<(int)(ML * DTK / 256), 256, 0, stream>>>(dblf, dtrf);
  dtprep_kernel<<<(int)(ML * DTK / 256), 256, 0, stream>>>(dblb, dtrb);

  wmma_gemm_kernel<DTK, false, false><<<GEMM_GRID((int)ML, DIN), 256, 0, stream>>>(
      dtrf, Bdf, dtwf, nullptr, (int)ML, DIN);                  // dt raw fwd
  wmma_gemm_kernel<DTK, false, false><<<GEMM_GRID((int)ML, DIN), 256, 0, stream>>>(
      dtrb, Bdb, dtwb, nullptr, (int)ML, DIN);                  // dt raw bwd

  ssm_scan_kernel<<<32, 256, 0, stream>>>(xc_f, dtwf, dblf, dt_b,   A_log,   Dp,   y_f);
  ssm_scan_kernel<<<32, 256, 0, stream>>>(xc_b, dtwb, dblb, dt_b_b, A_log_b, Dp_b, y_b);

  gate_kernel<<<ew_blocks, 256, 0, stream>>>(y_f, y_b, zbuf, ygh);

  // out = yg @ out_w^T + out_b, stored directly as (B, C, H, W)
  wmma_gemm_kernel<DIN, true, true><<<GEMM_GRID((int)ML, DMODEL), 256, 0, stream>>>(
      ygh, Bo, (float*)d_out, out_b, (int)ML, DMODEL);

  // second tuple element: x2 passthrough
  hipMemcpyAsync((float*)d_out + (size_t)BATCHN * DMODEL * LSEQ, x2,
                 (size_t)BATCHN * DMODEL * LSEQ * sizeof(float),
                 hipMemcpyDeviceToDevice, stream);
}
